// PrototypeStore_19894288515598
// MI455X (gfx1250) — compile-verified, matrix operand order checked by reading, not run on hardware
//
#include <hip/hip_runtime.h>

// Problem constants (match reference)
#define D_DIM   128
#define K_PROT  1024
#define MOM     0.9f

#define CHUNK_ROWS 64           // prototype rows per LDS buffer (double-buffered)
#define PROW       136          // padded halves per proto row in LDS (bank spread)
#define N_CHUNKS   (K_PROT / CHUNK_ROWS)   // 16

typedef _Float16 v8h  __attribute__((ext_vector_type(8)));
typedef _Float16 v16h __attribute__((ext_vector_type(16)));
typedef float    v8f  __attribute__((ext_vector_type(8)));

__device__ __forceinline__ void atomic_add_f32(float* p, float v) {
  __hip_atomic_fetch_add(p, v, __ATOMIC_RELAXED, __HIP_MEMORY_SCOPE_AGENT);
}

// ---------------------------------------------------------------------------
// Zero sums + counts region of workspace
// ---------------------------------------------------------------------------
__global__ void zero_kernel(float* __restrict__ p, int n) {
  int i = blockIdx.x * blockDim.x + threadIdx.x;
  if (i < n) p[i] = 0.0f;
}

// ---------------------------------------------------------------------------
// L2-normalize prototypes -> f16 (one block of 128 threads per prototype row)
// ---------------------------------------------------------------------------
__global__ void proto_norm_kernel(const float* __restrict__ protos,
                                  _Float16* __restrict__ pn) {
  const int k = blockIdx.x;
  const int t = threadIdx.x;        // 0..127
  float x = protos[k * D_DIM + t];
  float s = x * x;
#pragma unroll
  for (int m = 1; m < 32; m <<= 1) s += __shfl_xor(s, m, 32);
  __shared__ float red[4];
  if ((t & 31) == 0) red[t >> 5] = s;
  __syncthreads();
  const float n2 = red[0] + red[1] + red[2] + red[3];
  const float inv = 1.0f / fmaxf(sqrtf(n2), 1e-12f);
  pn[k * D_DIM + t] = (_Float16)(x * inv);
}

// ---------------------------------------------------------------------------
// Async DMA one 64-row prototype chunk (f16, row-major) into a padded LDS
// buffer using CDNA5 GLOBAL_LOAD_ASYNC_TO_LDS_B128 (ASYNCcnt-tracked, no
// VGPR staging). GVS addressing: SGPR base + 32-bit VGPR byte offset.
// Each wave issues 4 instructions x 32 lanes x 16B = 2 KB; 8 waves cover the
// full 16 KB chunk.
// ---------------------------------------------------------------------------
__device__ __forceinline__ void stage_chunk_async(const _Float16* __restrict__ pn,
                                                  int chunk, _Float16* buf,
                                                  int wave, int lane) {
  const unsigned ldsbase =
      (unsigned)(unsigned long long)(__attribute__((address_space(3))) void*)buf;
#pragma unroll
  for (int i = 0; i < 4; ++i) {
    const int idx = (wave * 4 + i) * 32 + lane;   // 0..1023 (uint4 index)
    const int row = idx >> 4;                      // 16 uint4 per 128-half row
    const int q   = idx & 15;
    const unsigned lofs = ldsbase + (unsigned)(row * (PROW * 2) + q * 16);
    const unsigned gofs = (unsigned)(chunk * CHUNK_ROWS * D_DIM * 2 + idx * 16);
    asm volatile("global_load_async_to_lds_b128 %0, %1, %2"
                 :
                 : "v"(lofs), "v"(gofs), "s"(pn)
                 : "memory");
  }
}

__device__ __forceinline__ void wait_async_all() {
  asm volatile("s_wait_asynccnt 0x0" ::: "memory");
}

// ---------------------------------------------------------------------------
// Fused: argmax(emb . proto_n) via WMMA  +  atomic segment sums/counts.
// Block = 256 threads (8 wave32). Each wave owns 16 embedding rows.
// A-matrix layout (16-bit, 16x32): lane(m,hi) element h -> K = (h/8)*16 + h%8 + hi*8
// B-matrix layout (32x16): lane(n,hi) same K mapping, column n = prototype row n.
// Both map to two contiguous 16-byte accesses from row-major storage.
// Prototype chunks are double-buffered in LDS with async DMA prefetch.
// ---------------------------------------------------------------------------
__global__ __launch_bounds__(256)
void assign_accum_kernel(const float* __restrict__ emb,
                         const _Float16* __restrict__ pn,
                         float* __restrict__ sums,
                         float* __restrict__ counts) {
  __shared__ __align__(16) _Float16 sp[2][CHUNK_ROWS * PROW];

  const int tid  = threadIdx.x;
  const int lane = tid & 31;
  const int wave = tid >> 5;
  const int m    = lane & 15;   // A row within tile / B column within tile
  const int hi   = lane >> 4;   // lane-half selects K sub-range
  const int rowbase = blockIdx.x * 128 + wave * 16;

  // Kick off DMA of the first prototype chunk before touching embeddings.
  stage_chunk_async(pn, 0, &sp[0][0], wave, lane);

  // ---- Load this wave's 16 embedding rows as WMMA A fragments (f32 -> f16).
  // No normalization needed: argmax is invariant to positive row scaling.
  v16h a[4];
  {
    const float* ap = emb + (size_t)(rowbase + m) * D_DIM + hi * 8;
#pragma unroll
    for (int c = 0; c < 4; ++c) {
      const float4 f0 = *(const float4*)(ap + c * 32 + 0);
      const float4 f1 = *(const float4*)(ap + c * 32 + 4);
      const float4 f2 = *(const float4*)(ap + c * 32 + 16);
      const float4 f3 = *(const float4*)(ap + c * 32 + 20);
      v16h av;
      av[0]  = (_Float16)f0.x; av[1]  = (_Float16)f0.y;
      av[2]  = (_Float16)f0.z; av[3]  = (_Float16)f0.w;
      av[4]  = (_Float16)f1.x; av[5]  = (_Float16)f1.y;
      av[6]  = (_Float16)f1.z; av[7]  = (_Float16)f1.w;
      av[8]  = (_Float16)f2.x; av[9]  = (_Float16)f2.y;
      av[10] = (_Float16)f2.z; av[11] = (_Float16)f2.w;
      av[12] = (_Float16)f3.x; av[13] = (_Float16)f3.y;
      av[14] = (_Float16)f3.z; av[15] = (_Float16)f3.w;
      a[c] = av;
    }
  }

  float    maxv[8];
  unsigned maxi[8];
#pragma unroll
  for (int r = 0; r < 8; ++r) { maxv[r] = -3.4e38f; maxi[r] = 0u; }

  // First chunk must be resident before compute.
  wait_async_all();
  __syncthreads();

  // ---- Sweep all prototypes; prefetch chunk+1 while computing chunk.
  for (int chunk = 0; chunk < N_CHUNKS; ++chunk) {
    const _Float16* cur = &sp[chunk & 1][0];
    _Float16*       nxt = &sp[(chunk + 1) & 1][0];
    if (chunk + 1 < N_CHUNKS)
      stage_chunk_async(pn, chunk + 1, nxt, wave, lane);

#pragma unroll
    for (int t = 0; t < CHUNK_ROWS / 16; ++t) {    // 16-prototype tiles
      const _Float16* bp = cur + (t * 16 + m) * PROW + hi * 8;
      v8f acc = {0.f, 0.f, 0.f, 0.f, 0.f, 0.f, 0.f, 0.f};
#pragma unroll
      for (int c = 0; c < 4; ++c) {                // K = 128 in 4 x 32 steps
        const v8h blo = *(const v8h*)(bp + c * 32);
        const v8h bhi = *(const v8h*)(bp + c * 32 + 16);
        const v16h b = __builtin_shufflevector(
            blo, bhi, 0, 1, 2, 3, 4, 5, 6, 7, 8, 9, 10, 11, 12, 13, 14, 15);
        acc = __builtin_amdgcn_wmma_f32_16x16x32_f16(
            false, a[c], false, b, (short)0, acc, false, false);
      }
      const unsigned col = (unsigned)(chunk * CHUNK_ROWS + t * 16 + m);
#pragma unroll
      for (int r = 0; r < 8; ++r) {
        if (acc[r] > maxv[r]) { maxv[r] = acc[r]; maxi[r] = col; }
      }
    }

    // Next buffer filled (own ASYNCcnt) + everyone done reading `cur`.
    wait_async_all();
    __syncthreads();
  }

  // ---- Argmax reduction across the 16 column-residue lanes (wave32 halves).
#pragma unroll
  for (int r = 0; r < 8; ++r) {
    float    v = maxv[r];
    unsigned i = maxi[r];
#pragma unroll
    for (int s = 1; s < 16; s <<= 1) {
      const float    ov = __shfl_xor(v, s, 32);
      const unsigned oi = (unsigned)__shfl_xor((int)i, s, 32);
      if (ov > v || (ov == v && oi < i)) { v = ov; i = oi; }
    }
    maxi[r] = i;  // replicated across each 16-lane half
  }

  // ---- Atomic segment sums / counts (raw embeddings).
  // C/D layout: VGPR r -> rows rowbase+r (lanes 0-15) and rowbase+r+8 (16-31).
  for (int rr = 0; rr < 16; ++rr) {
    const int r = rr & 7;
    const unsigned aidx =
        (unsigned)__shfl((int)maxi[r], (rr < 8) ? 0 : 16, 32);
    const float* erow = emb + (size_t)(rowbase + rr) * D_DIM;
    float* srow = sums + (size_t)aidx * D_DIM;
#pragma unroll
    for (int j = 0; j < 4; ++j)
      atomic_add_f32(srow + lane + j * 32, erow[lane + j * 32]);
    if (lane == 0) atomic_add_f32(counts + aidx, 1.0f);
  }
}

// ---------------------------------------------------------------------------
// EMA update: out = counts>0 ? 0.9*proto + 0.1*(sums/max(counts,1)) : proto
// ---------------------------------------------------------------------------
__global__ void ema_kernel(const float* __restrict__ protos,
                           const float* __restrict__ sums,
                           const float* __restrict__ counts,
                           float* __restrict__ out) {
  const int k = blockIdx.x;
  const int d = threadIdx.x;
  const float c = counts[k];
  const float p = protos[k * D_DIM + d];
  const float mean = sums[k * D_DIM + d] / fmaxf(c, 1.0f);
  out[k * D_DIM + d] = (c > 0.0f) ? (MOM * p + (1.0f - MOM) * mean) : p;
}

// ---------------------------------------------------------------------------
extern "C" void kernel_launch(void* const* d_in, const int* in_sizes, int n_in,
                              void* d_out, int out_size, void* d_ws,
                              size_t ws_size, hipStream_t stream) {
  (void)n_in; (void)out_size; (void)ws_size;
  const float* emb    = (const float*)d_in[0];   // (B, 128) f32
  const float* protos = (const float*)d_in[1];   // (1024, 128) f32
  float* out = (float*)d_out;                    // (1024, 128) f32

  const int B = in_sizes[0] / D_DIM;             // 262144

  // Workspace layout: [proto_f16: K*D*2][sums: K*D*4][counts: K*4]
  char* ws = (char*)d_ws;
  _Float16* pn   = (_Float16*)ws;
  float*    sums = (float*)(ws + (size_t)K_PROT * D_DIM * sizeof(_Float16));
  float* counts  = sums + (size_t)K_PROT * D_DIM;   // contiguous after sums

  const int nzero = K_PROT * D_DIM + K_PROT;        // sums + counts together
  zero_kernel<<<(nzero + 255) / 256, 256, 0, stream>>>(sums, nzero);
  proto_norm_kernel<<<K_PROT, D_DIM, 0, stream>>>(protos, pn);
  assign_accum_kernel<<<B / 128, 256, 0, stream>>>(emb, pn, sums, counts);
  ema_kernel<<<K_PROT, D_DIM, 0, stream>>>(protos, sums, counts, out);
}
